// Transducer_54649163875061
// MI455X (gfx1250) — matrix operand order repaired and checked by reading, not compiled, not run on hardware
//
#include <hip/hip_runtime.h>
#include <hip/hip_bf16.h>
#include <math.h>

// ---------------- problem constants ----------------
#define B_    16
#define T_    1024
#define HENC  512
#define E_    512
#define HDEC  640
#define J_    640
#define V_    4096
#define G4    2560   // 4*HDEC

// ---------------- launch shape ----------------
#define NBLK  32
#define NTHR  256
#define NWAVE (NBLK * (NTHR / 32))   // 256 waves total

typedef __attribute__((ext_vector_type(16))) __bf16 v16bf;
typedef __attribute__((ext_vector_type(8)))  __bf16 v8bf;
typedef __attribute__((ext_vector_type(8)))  float  v8f;

// ---------------- workspace layout (bytes) ----------------
#define ALIGN256(x) (((x) + 255) & ~(size_t)255)
static constexpr size_t OFF_CTRL  = 0;                                           // barrier counter
static constexpr size_t OFF_WENCT = 256;                                         // [J][HENC] bf16
static constexpr size_t OFF_WDECT = ALIGN256(OFF_WENCT + (size_t)J_ * HENC * 2); // [J][HDEC]
static constexpr size_t OFF_WOUTT = ALIGN256(OFF_WDECT + (size_t)J_ * HDEC * 2); // [V][J]
static constexpr size_t OFF_WXT   = ALIGN256(OFF_WOUTT + (size_t)V_ * J_ * 2);   // [G4][E]
static constexpr size_t OFF_WHT   = ALIGN256(OFF_WXT   + (size_t)G4 * E_ * 2);   // [G4][HDEC]
static constexpr size_t OFF_FRAME = ALIGN256(OFF_WHT   + (size_t)G4 * HDEC * 2); // [B][HENC] bf16
static constexpr size_t OFF_DECBF = ALIGN256(OFF_FRAME + (size_t)B_ * HENC * 2); // [B][HDEC] bf16 (h)
static constexpr size_t OFF_XBF   = ALIGN256(OFF_DECBF + (size_t)B_ * HDEC * 2); // [B][E]    bf16
static constexpr size_t OFF_ZBF   = ALIGN256(OFF_XBF   + (size_t)B_ * E_ * 2);   // [B][J]    bf16
static constexpr size_t OFF_C     = ALIGN256(OFF_ZBF   + (size_t)B_ * J_ * 2);   // [B][HDEC] f32
static constexpr size_t OFF_U     = ALIGN256(OFF_C     + (size_t)B_ * HDEC * 4); // [B][V]    f32
static constexpr size_t OFF_G     = ALIGN256(OFF_U     + (size_t)B_ * V_ * 4);   // [B][G4]   f32
static constexpr size_t OFF_PMAX  = ALIGN256(OFF_G     + (size_t)B_ * G4 * 4);   // [NWAVE][B] f32
static constexpr size_t OFF_PSUM  = ALIGN256(OFF_PMAX  + (size_t)NWAVE * B_ * 4);
static constexpr size_t OFF_PIDX  = ALIGN256(OFF_PSUM  + (size_t)NWAVE * B_ * 4);
static constexpr size_t OFF_LSE   = ALIGN256(OFF_PIDX  + (size_t)NWAVE * B_ * 4);
static constexpr size_t WS_NEED   = ALIGN256(OFF_LSE + B_ * 4);  // ~13.2 MB

// ---------------- helpers ----------------
static __device__ __forceinline__ unsigned short f2bfu(float x) {
  unsigned u = __float_as_uint(x);
  return (unsigned short)((u + 0x7FFFu + ((u >> 16) & 1u)) >> 16);   // RNE fp32->bf16
}
static __device__ __forceinline__ float sigm(float x) { return 1.0f / (1.0f + __expf(-x)); }

// Device-wide monotonic barrier (persistent kernel; all NBLK blocks resident).
static __device__ __forceinline__ void gsync(unsigned* cnt, unsigned& bar) {
  __syncthreads();
  if (threadIdx.x == 0) {
    bar += NBLK;
    __threadfence();
    __hip_atomic_fetch_add(cnt, 1u, __ATOMIC_ACQ_REL, __HIP_MEMORY_SCOPE_AGENT);
    while (__hip_atomic_load(cnt, __ATOMIC_ACQUIRE, __HIP_MEMORY_SCOPE_AGENT) < bar) {
      __builtin_amdgcn_s_sleep(2);
    }
    __threadfence();
  }
  __syncthreads();
}

// A fragment: activations row-major [M=16][K] bf16.
// ISA 16-bit A 16x32 layout: lane l (M=l&15, h=l>>4) holds K = k0+8h+{0..7} and k0+16+8h+{0..7}.
static __device__ __forceinline__ v16bf load_afrag(const unsigned short* act, int ld, int k0, int lane) {
  const int M = lane & 15, h = lane >> 4;
  const unsigned short* p = act + M * ld + k0 + 8 * h;
  v8bf lo = *(const v8bf*)(const void*)p;
  v8bf hi = *(const v8bf*)(const void*)(p + 16);
  v16bf a;
#pragma unroll
  for (int i = 0; i < 8; ++i) { a[i] = lo[i]; a[i + 8] = hi[i]; }
  return a;
}

// B fragment: weights pre-transposed row-major [N][K] bf16.
// ISA 16-bit B 32x16 layout: lane l (N=n0+(l&15), h=l>>4) holds K = k0+8h+{0..7} and k0+16+8h+{0..7}.
static __device__ __forceinline__ v16bf load_bfrag(const unsigned short* wt, int ldk, int n0, int k0, int lane) {
  const int N = n0 + (lane & 15), h = lane >> 4;
  const unsigned short* p = wt + (size_t)N * ldk + k0 + 8 * h;
  v8bf lo = *(const v8bf*)(const void*)p;
  v8bf hi = *(const v8bf*)(const void*)(p + 16);
  v16bf b;
#pragma unroll
  for (int i = 0; i < 8; ++i) { b[i] = lo[i]; b[i + 8] = hi[i]; }
  return b;
}

// acc += act[16 x kt] * wt[kt x 16 (cols n0..n0+15)]
static __device__ __forceinline__ void mma_panel(v8f& acc, const unsigned short* act, int ld,
                                                 const unsigned short* wt, int ldk,
                                                 int n0, int kt, int lane) {
  for (int k0 = 0; k0 < kt; k0 += 32) {
    v16bf a = load_afrag(act, ld, k0, lane);
    v16bf b = load_bfrag(wt, ldk, n0, k0, lane);
    acc = __builtin_amdgcn_wmma_f32_16x16x32_bf16(false, a, false, b, (short)0, acc, false, false);
  }
}

// LSTM gates: g[B][G4] = x@Wx + h@Wh + b_lstm (one 16-col tile per wave, 160 tiles)
static __device__ __forceinline__ void gates_phase(int wv, int lane,
    const unsigned short* x_bf, const unsigned short* dec_bf,
    const unsigned short* WxT, const unsigned short* WhT,
    const float* __restrict__ b_lstm, float* __restrict__ g) {
  if (wv >= G4 / 16) return;
  const int n0 = wv * 16;
  v8f acc = {};
  mma_panel(acc, x_bf,   E_,   WxT, E_,   n0, E_,   lane);
  mma_panel(acc, dec_bf, HDEC, WhT, HDEC, n0, HDEC, lane);
  const int col = n0 + (lane & 15), h = lane >> 4;
  const float bb = b_lstm[col];
#pragma unroll
  for (int r = 0; r < 8; ++r) g[(size_t)(r + 8 * h) * G4 + col] = acc[r] + bb;
}

// Elementwise LSTM update: c' = sig(f)*c + sig(i)*tanh(g); h' = sig(o)*tanh(c'); dec_bf = bf16(h')
static __device__ __forceinline__ void lstm_update(int gtid, const float* __restrict__ g,
                                                   float* __restrict__ c, unsigned short* dec_bf) {
  for (int i = gtid; i < B_ * HDEC; i += NBLK * NTHR) {
    const int b = i / HDEC, j = i - b * HDEC;
    const float* gb = g + (size_t)b * G4;
    const float ii = gb[j], ff = gb[j + HDEC], gg = gb[j + 2 * HDEC], oo = gb[j + 3 * HDEC];
    const float cc = sigm(ff) * c[i] + sigm(ii) * tanhf(gg);
    c[i] = cc;
    dec_bf[i] = f2bfu(sigm(oo) * tanhf(cc));
  }
}

// ---------------- prep: fp32 [K][N] -> bf16 [N][K] ----------------
__global__ void k_transpose_bf16(const float* __restrict__ src, unsigned short* __restrict__ dst,
                                 int K, int N) {
  const long total = (long)K * N;
  for (long i = (long)blockIdx.x * blockDim.x + threadIdx.x; i < total;
       i += (long)gridDim.x * blockDim.x) {
    const int n = (int)(i / K), k = (int)(i % K);
    dst[i] = f2bfu(src[(long)k * N + n]);
  }
}

// ---------------- main persistent kernel ----------------
__global__ __launch_bounds__(NTHR, 1)
void transducer_main(const float* __restrict__ enc, const int* __restrict__ lengths,
                     const float* __restrict__ embed,
                     const float* __restrict__ b_lstm, const float* __restrict__ b_j,
                     const float* __restrict__ b_out,
                     float* __restrict__ out_logp, float* __restrict__ out_tok,
                     char* __restrict__ ws) {
  unsigned* cnt = (unsigned*)(ws + OFF_CTRL);
  const unsigned short* WencT = (const unsigned short*)(ws + OFF_WENCT);
  const unsigned short* WdecT = (const unsigned short*)(ws + OFF_WDECT);
  const unsigned short* WoutT = (const unsigned short*)(ws + OFF_WOUTT);
  const unsigned short* WxT   = (const unsigned short*)(ws + OFF_WXT);
  const unsigned short* WhT   = (const unsigned short*)(ws + OFF_WHT);
  unsigned short* frame_bf = (unsigned short*)(ws + OFF_FRAME);
  unsigned short* dec_bf   = (unsigned short*)(ws + OFF_DECBF);
  unsigned short* x_bf     = (unsigned short*)(ws + OFF_XBF);
  unsigned short* z_bf     = (unsigned short*)(ws + OFF_ZBF);
  float* c_st  = (float*)(ws + OFF_C);
  float* u_st  = (float*)(ws + OFF_U);
  float* g_st  = (float*)(ws + OFF_G);
  float* pmax  = (float*)(ws + OFF_PMAX);
  float* psum  = (float*)(ws + OFF_PSUM);
  int*   pidx  = (int*)  (ws + OFF_PIDX);
  float* lse   = (float*)(ws + OFF_LSE);

  const int lane = threadIdx.x & 31;
  const int wv   = blockIdx.x * (NTHR / 32) + (threadIdx.x >> 5);
  const int gtid = blockIdx.x * NTHR + threadIdx.x;
  unsigned bar = 0;
  __shared__ int spred[B_];

  // ---------- init: h=c=0, x = embed[BOS=0] ----------
  for (int i = gtid; i < B_ * HDEC; i += NBLK * NTHR) { c_st[i] = 0.0f; dec_bf[i] = 0; }
  for (int i = gtid; i < B_ * E_;   i += NBLK * NTHR) {
    x_bf[i] = f2bfu(embed[i % E_]);   // row 0 broadcast across batch
  }
  gsync(cnt, bar);
  gates_phase(wv, lane, x_bf, dec_bf, WxT, WhT, b_lstm, g_st);
  gsync(cnt, bar);
  lstm_update(gtid, g_st, c_st, dec_bf);   // h0 -> dec_bf, c0 -> c_st
  gsync(cnt, bar);

  // ---------- sequential decode over T ----------
  for (int t = 0; t < T_; ++t) {
    // S: stage frame_t as bf16
    for (int i = gtid; i < B_ * HENC; i += NBLK * NTHR) {
      const int b = i >> 9, k = i & (HENC - 1);
      frame_bf[i] = f2bfu(enc[((size_t)b * T_ + t) * HENC + k]);
    }
    gsync(cnt, bar);

    // Z: z = tanh(frame@W_enc + dec@W_dec + b_j), 40 tiles
    if (wv < J_ / 16) {
      const int n0 = wv * 16;
      v8f acc = {};
      mma_panel(acc, frame_bf, HENC, WencT, HENC, n0, HENC, lane);
      mma_panel(acc, dec_bf,   HDEC, WdecT, HDEC, n0, HDEC, lane);
      const int col = n0 + (lane & 15), h = lane >> 4;
      const float bj = b_j[col];
#pragma unroll
      for (int r = 0; r < 8; ++r)
        z_bf[(size_t)(r + 8 * h) * J_ + col] = f2bfu(tanhf(acc[r] + bj));
    }
    gsync(cnt, bar);

    // U: u = z@W_out + b_out, 256 tiles (one per wave) + online max/argmax/sumexp partials
    {
      const int n0 = wv * 16;
      v8f acc = {};
      mma_panel(acc, z_bf, J_, WoutT, J_, n0, J_, lane);
      const int col = n0 + (lane & 15), h = lane >> 4;
      const float bo = b_out[col];
      float vals[8];
#pragma unroll
      for (int r = 0; r < 8; ++r) {
        const float v = acc[r] + bo;
        vals[r] = v;
        u_st[(size_t)(r + 8 * h) * V_ + col] = v;
      }
#pragma unroll
      for (int r = 0; r < 8; ++r) {
        float m = vals[r]; int id = col;
#pragma unroll
        for (int off = 1; off < 16; off <<= 1) {
          const float om = __shfl_xor(m, off, 32);
          const int   oi = __shfl_xor(id, off, 32);
          if (om > m || (om == m && oi < id)) { m = om; id = oi; }
        }
        float s = __expf(vals[r] - m);
#pragma unroll
        for (int off = 1; off < 16; off <<= 1) s += __shfl_xor(s, off, 32);
        if ((lane & 15) == 0) {
          const int row = r + 8 * h;
          pmax[wv * B_ + row] = m;
          pidx[wv * B_ + row] = id;
          psum[wv * B_ + row] = s;
        }
      }
    }
    gsync(cnt, bar);

    // R: global reduce -> lse, pred; token write; x_bf = bf16(embed[pred])
    if (blockIdx.x == 0) {
      if (threadIdx.x < B_) {
        const int b = threadIdx.x;
        float m = -INFINITY; int id = 0;
        for (int w = 0; w < NWAVE; ++w) {
          const float pm = pmax[w * B_ + b];
          if (pm > m) { m = pm; id = pidx[w * B_ + b]; }
        }
        float s = 0.0f;
        for (int w = 0; w < NWAVE; ++w)
          s += psum[w * B_ + b] * __expf(pmax[w * B_ + b] - m);
        lse[b] = m + __logf(s);
        const int pr = (t < lengths[b]) ? id : 0;
        spred[b] = pr;
        out_tok[(size_t)t * B_ + b] = (float)pr;
      }
      __syncthreads();
      for (int i = threadIdx.x; i < B_ * E_; i += NTHR) {
        const int b = i / E_, k = i - b * E_;
        x_bf[i] = f2bfu(embed[(size_t)spred[b] * E_ + k]);
      }
    }
    gsync(cnt, bar);

    // G: LSTM gates (160 tiles)
    gates_phase(wv, lane, x_bf, dec_bf, WxT, WhT, b_lstm, g_st);
    gsync(cnt, bar);

    // E: LSTM elementwise update + logp = u - lse
    lstm_update(gtid, g_st, c_st, dec_bf);
    for (int i = gtid; i < B_ * V_; i += NBLK * NTHR) {
      out_logp[(size_t)t * (B_ * V_) + i] = u_st[i] - lse[i >> 12];
    }
    gsync(cnt, bar);
  }
}

// ---------------- host launch ----------------
extern "C" void kernel_launch(void* const* d_in, const int* in_sizes, int n_in,
                              void* d_out, int out_size, void* d_ws, size_t ws_size,
                              hipStream_t stream) {
  (void)in_sizes; (void)n_in; (void)out_size; (void)ws_size;
  const float* enc     = (const float*)d_in[0];
  const int*   lengths = (const int*)  d_in[1];
  const float* embed   = (const float*)d_in[2];
  const float* Wx      = (const float*)d_in[3];
  const float* Wh      = (const float*)d_in[4];
  const float* b_lstm  = (const float*)d_in[5];
  const float* W_enc   = (const float*)d_in[6];
  const float* W_dec   = (const float*)d_in[7];
  const float* b_j     = (const float*)d_in[8];
  const float* W_out   = (const float*)d_in[9];
  const float* b_out   = (const float*)d_in[10];

  char* ws = (char*)d_ws;
  float* out_logp = (float*)d_out;
  float* out_tok  = out_logp + (size_t)T_ * B_ * V_;

  // zero the device barrier counter (capturable async memset node)
  hipMemsetAsync(ws + OFF_CTRL, 0, 256, stream);

  // one-time weight transpose + bf16 convert (weights then live in L2 across the whole decode)
  k_transpose_bf16<<<512, 256, 0, stream>>>(W_enc, (unsigned short*)(ws + OFF_WENCT), HENC, J_);
  k_transpose_bf16<<<512, 256, 0, stream>>>(W_dec, (unsigned short*)(ws + OFF_WDECT), HDEC, J_);
  k_transpose_bf16<<<1024, 256, 0, stream>>>(W_out, (unsigned short*)(ws + OFF_WOUTT), J_, V_);
  k_transpose_bf16<<<512, 256, 0, stream>>>(Wx,    (unsigned short*)(ws + OFF_WXT),   E_, G4);
  k_transpose_bf16<<<1024, 256, 0, stream>>>(Wh,   (unsigned short*)(ws + OFF_WHT),   HDEC, G4);

  // persistent sequential decoder
  transducer_main<<<NBLK, NTHR, 0, stream>>>(enc, lengths, embed, b_lstm, b_j, b_out,
                                             out_logp, out_tok, ws);
}
